// DACRVQVAEBottleneck_44298292691486
// MI455X (gfx1250) — compile-verified, matrix operand order checked by reading, not run on hardware
//
#include <hip/hip_runtime.h>
#include <hip/hip_bf16.h>

typedef __attribute__((ext_vector_type(16))) _Float16 v16h;
typedef __attribute__((ext_vector_type(8)))  float    v8f;

#define NC 9
#define CS 1024
#define CD 8
#define DIM 128
#define BB 16
#define TLEN 4096
#define TILE 128
#define NTHREADS 256

// d_out flat layout (reference return order): z_q, kl, commitment, codebook, codes, latents
#define ZQ_OFF    ((size_t)0)
#define KL_OFF    ((size_t)BB*DIM*TLEN)
#define CM_OFF    (KL_OFF + 1)
#define CBL_OFF   (KL_OFF + 2)
#define CODES_OFF (KL_OFF + 3)
#define LAT_OFF   (CODES_OFF + (size_t)BB*NC*TLEN)

#define RES_STRIDE 132        // float4-aligned LDS row stride (2-way conflict max)
#define CBROW 3               // codebook row stride in uint4 units (48B: conflict-free)

// ---------------- kernel: zero the 3 scalar accumulators ----------------
__global__ void rvq_zero_scalars(float* __restrict__ out) {
    if (threadIdx.x < 3) out[KL_OFF + threadIdx.x] = 0.0f;
}

// ---------------- kernel: precompute L2-normalized codebook as packed f16 ----------------
__global__ void rvq_norm_cb(const float* __restrict__ cb, uint4* __restrict__ wsn) {
    int r = blockIdx.x * blockDim.x + threadIdx.x;      // row 0..NC*CS-1
    if (r >= NC * CS) return;
    const float* p = cb + (size_t)r * CD;
    float v[8]; float ss = 0.0f;
#pragma unroll
    for (int j = 0; j < 8; j++) { v[j] = p[j]; ss += v[j] * v[j]; }
    float inv = 1.0f / fmaxf(sqrtf(ss), 1e-12f);
    union { uint4 u; _Float16 h[8]; } o;
#pragma unroll
    for (int j = 0; j < 8; j++) o.h[j] = (_Float16)(v[j] * inv);
    wsn[r] = o.u;
}

// ---------------- main fused kernel ----------------
__launch_bounds__(NTHREADS, 1)
__global__ void rvq_main(const float* __restrict__ x,     const float* __restrict__ noise,
                         const float* __restrict__ in_w,  const float* __restrict__ in_b,
                         const float* __restrict__ cb_raw,const float* __restrict__ out_w,
                         const float* __restrict__ out_b, const uint4* __restrict__ cbn,
                         float* __restrict__ out)
{
    __shared__ float s_res[TILE * RES_STRIDE];     // residual [tok][d]
    __shared__ float s_ze [TILE * 9];              // z_e [tok][c]
    __shared__ float s_zq [TILE * 9];              // gathered raw codebook vectors
    __shared__ uint4 s_cbn4[CBROW * CS + CBROW];   // f16 rows @48B stride; odd quads zero; zero row
    __shared__ float s_inw [CD * DIM];
    __shared__ float s_outw[DIM * CD];
    __shared__ float s_inb [CD];
    __shared__ float s_outb[DIM];
    __shared__ int   s_idx [TILE];
    __shared__ float s_red [8];

    const int tid = threadIdx.x;
    const int blk = blockIdx.x;
    const int b   = blk >> 5;
    const int t0  = (blk & 31) * TILE;
    const int tok = tid & (TILE - 1);
    const int dg  = tid >> 7;
    const int dbase = dg * 64;
    const int wv  = tid >> 5;
    const int ln  = tid & 31;

    float klacc   = 0.0f;
    float lossacc = 0.0f;

    // zero-fill codebook staging region once (pad quads + zero row stay zero forever)
    for (int j = tid; j < CBROW * CS + CBROW; j += NTHREADS)
        s_cbn4[j] = make_uint4(0u, 0u, 0u, 0u);

    // ---- phase 0: VAE sample; z -> LDS residual + z_q slot in global; KL partial ----
    const float* xm = x     + (size_t)b * 2 * DIM * TLEN + t0;
    const float* xs = x     + (size_t)b * 2 * DIM * TLEN + (size_t)DIM * TLEN + t0;
    const float* nb = noise + (size_t)b * DIM * TLEN + t0;
    float* zo = out + ZQ_OFF + (size_t)b * DIM * TLEN + (size_t)dbase * TLEN + t0 + tok;
#pragma unroll 4
    for (int k = 0; k < 64; k++) {
        int d = dbase + k;
        float m  = xm[(size_t)d * TLEN + tok];
        float sc = xs[(size_t)d * TLEN + tok];
        float nz = nb[(size_t)d * TLEN + tok];
        float sp = fmaxf(sc, 0.0f) + __logf(1.0f + __expf(-fabsf(sc)));   // stable softplus
        float sd = sp + 1e-4f;
        float var = sd * sd;
        klacc += m * m + var - 2.0f * __logf(sd) - 1.0f;
        float z = nz * sd + m;
        s_res[tok * RES_STRIDE + d] = z;
        zo[(size_t)k * TLEN] = z;            // stash z; epilogue computes z_q = z - residual
    }
    __syncthreads();

    // ---- 9 sequential codebooks, residual stays in LDS ----
    for (int i = 0; i < NC; i++) {
        // stage weights + normalized codebook into LDS
        {
            const float4* iw4 = (const float4*)(in_w  + (size_t)i * CD * DIM);
            const float4* ow4 = (const float4*)(out_w + (size_t)i * DIM * CD);
            ((float4*)s_inw )[tid] = iw4[tid];
            ((float4*)s_outw)[tid] = ow4[tid];
            if (tid < CD)  s_inb [tid] = in_b [i * CD  + tid];
            if (tid < DIM) s_outb[tid] = out_b[i * DIM + tid];
#pragma unroll
            for (int j = 0; j < 4; j++) {
                int r = tid + j * NTHREADS;
                s_cbn4[CBROW * r] = cbn[i * CS + r];
            }
        }
        __syncthreads();

        // in_proj: z_e = in_w @ residual + in_b   (1 token, 4 channels per thread)
        {
            const int cbase = dg * 4;
            float acc[4] = {0.f, 0.f, 0.f, 0.f};
            const float* rp = s_res + tok * RES_STRIDE;
            for (int d = 0; d < DIM; d += 4) {
                float4 r4 = *(const float4*)(rp + d);
#pragma unroll
                for (int j = 0; j < 4; j++) {
                    float4 w = *(const float4*)(s_inw + (cbase + j) * DIM + d);
                    acc[j] = fmaf(w.x, r4.x, acc[j]);
                    acc[j] = fmaf(w.y, r4.y, acc[j]);
                    acc[j] = fmaf(w.z, r4.z, acc[j]);
                    acc[j] = fmaf(w.w, r4.w, acc[j]);
                }
            }
#pragma unroll
            for (int j = 0; j < 4; j++) {
                float ze = acc[j] + s_inb[cbase + j];
                s_ze[tok * 9 + cbase + j] = ze;
                out[LAT_OFF + (size_t)b * NC * CD * TLEN
                            + (size_t)(i * CD + cbase + j) * TLEN + t0 + tok] = ze;
            }
        }
        __syncthreads();

        // codebook search: per wave, 16 tokens; argmax of enc_n . cb_n via WMMA
        {
            // A fragment (16x32 f16): lanes 0-15 carry K=0..7, everything else zero
            v16h afrag = {};
            if (ln < 16) {
                int mytok = wv * 16 + ln;
                float e[8]; float ss = 0.0f;
#pragma unroll
                for (int c = 0; c < 8; c++) { e[c] = s_ze[mytok * 9 + c]; ss += e[c] * e[c]; }
                float inv = 1.0f / fmaxf(sqrtf(ss), 1e-12f);
#pragma unroll
                for (int c = 0; c < 8; c++) afrag[c] = (_Float16)(e[c] * inv);
            }

            // C bias = +2.0 -> scores positive -> raw f32 bits are uint-monotonic
            v8f ctwo;
#pragma unroll
            for (int r = 0; r < 8; r++) ctwo[r] = 2.0f;

            unsigned key[8];
#pragma unroll
            for (int r = 0; r < 8; r++) key[r] = 0u;

            // per-lane B pointer: lanes 0-15 walk the codebook; lanes 16-31 pinned to zero row
            const uint4* bp  = s_cbn4 + ((ln < 16) ? (CBROW * ln) : (CBROW * CS));
            const int pstep  = (ln < 16) ? (CBROW * 16) : 0;
            unsigned nbase   = (unsigned)(ln & 15);
            union U16 { uint4 q[2]; v16h h; };

            for (int jt = 0; jt < CS / 16; jt += 4) {
                U16 u0, u1, u2, u3;
                u0.q[0] = bp[0];             u0.q[1] = bp[1];              // data + zero quad
                u1.q[0] = bp[pstep];         u1.q[1] = bp[pstep + 1];
                u2.q[0] = bp[2 * pstep];     u2.q[1] = bp[2 * pstep + 1];
                u3.q[0] = bp[3 * pstep];     u3.q[1] = bp[3 * pstep + 1];
                bp += 4 * pstep;

                v8f c0 = __builtin_amdgcn_wmma_f32_16x16x32_f16(false, afrag, false, u0.h, (short)0, ctwo, false, false);
                v8f c1 = __builtin_amdgcn_wmma_f32_16x16x32_f16(false, afrag, false, u1.h, (short)0, ctwo, false, false);
                v8f c2 = __builtin_amdgcn_wmma_f32_16x16x32_f16(false, afrag, false, u2.h, (short)0, ctwo, false, false);
                v8f c3 = __builtin_amdgcn_wmma_f32_16x16x32_f16(false, afrag, false, u3.h, (short)0, ctwo, false, false);

                unsigned n0 = nbase, n1 = nbase + 16u, n2 = nbase + 32u, n3 = nbase + 48u;
#pragma unroll
                for (int r = 0; r < 8; r++) {
                    unsigned k0 = (__float_as_uint(c0[r]) & 0xFFFFFC00u) | n0;
                    unsigned k1 = (__float_as_uint(c1[r]) & 0xFFFFFC00u) | n1;
                    unsigned k2 = (__float_as_uint(c2[r]) & 0xFFFFFC00u) | n2;
                    unsigned k3 = (__float_as_uint(c3[r]) & 0xFFFFFC00u) | n3;
                    key[r] = key[r] > k0 ? key[r] : k0;
                    key[r] = key[r] > k1 ? key[r] : k1;
                    key[r] = key[r] > k2 ? key[r] : k2;
                    key[r] = key[r] > k3 ? key[r] : k3;
                }
                nbase += 64u;
            }

            // butterfly argmax over the 16 lanes sharing a token half: single v_max_u32 lane
#pragma unroll
            for (int off = 1; off < 16; off <<= 1) {
#pragma unroll
                for (int r = 0; r < 8; r++) {
                    unsigned ok = (unsigned)__shfl_xor((int)key[r], off, 16);
                    key[r] = key[r] > ok ? key[r] : ok;
                }
            }
            if (ln == 0 || ln == 16) {
                int base = wv * 16 + ((ln == 16) ? 8 : 0);
#pragma unroll
                for (int r = 0; r < 8; r++) {
                    int idx = (int)(key[r] & 0x3FFu);
                    s_idx[base + r] = idx;
                    out[CODES_OFF + (size_t)b * NC * TLEN + (size_t)i * TLEN
                                  + t0 + base + r] = (float)idx;
                }
            }
        }
        __syncthreads();

        // gather raw codebook rows + loss partials
        {
            int gt = tid >> 1;
            int c0 = (tid & 1) * 4;
            int idx = s_idx[gt];
            const float4 v = *(const float4*)(cb_raw + ((size_t)i * CS + idx) * CD + c0);
            float d0 = s_ze[gt * 9 + c0 + 0] - v.x;
            float d1 = s_ze[gt * 9 + c0 + 1] - v.y;
            float d2 = s_ze[gt * 9 + c0 + 2] - v.z;
            float d3 = s_ze[gt * 9 + c0 + 3] - v.w;
            lossacc += d0 * d0 + d1 * d1 + d2 * d2 + d3 * d3;
            s_zq[gt * 9 + c0 + 0] = v.x;
            s_zq[gt * 9 + c0 + 1] = v.y;
            s_zq[gt * 9 + c0 + 2] = v.z;
            s_zq[gt * 9 + c0 + 3] = v.w;
        }
        __syncthreads();

        // out_proj + residual update: residual -= out_w @ zq + out_b
        {
            float zq[8];
#pragma unroll
            for (int c = 0; c < 8; c++) zq[c] = s_zq[tok * 9 + c];
            float* rp = s_res + tok * RES_STRIDE + dbase;
#pragma unroll 4
            for (int k = 0; k < 64; k += 4) {
                int d = dbase + k;
                float4 r4 = *(const float4*)(rp + k);
                float4 ob = *(const float4*)(s_outb + d);
                const float4* wp = (const float4*)(s_outw + d * 8);
                float o0 = ob.x, o1 = ob.y, o2 = ob.z, o3 = ob.w;
                float4 wa, wb;
                wa = wp[0]; wb = wp[1];
                o0 = fmaf(wa.x,zq[0],o0); o0 = fmaf(wa.y,zq[1],o0); o0 = fmaf(wa.z,zq[2],o0); o0 = fmaf(wa.w,zq[3],o0);
                o0 = fmaf(wb.x,zq[4],o0); o0 = fmaf(wb.y,zq[5],o0); o0 = fmaf(wb.z,zq[6],o0); o0 = fmaf(wb.w,zq[7],o0);
                wa = wp[2]; wb = wp[3];
                o1 = fmaf(wa.x,zq[0],o1); o1 = fmaf(wa.y,zq[1],o1); o1 = fmaf(wa.z,zq[2],o1); o1 = fmaf(wa.w,zq[3],o1);
                o1 = fmaf(wb.x,zq[4],o1); o1 = fmaf(wb.y,zq[5],o1); o1 = fmaf(wb.z,zq[6],o1); o1 = fmaf(wb.w,zq[7],o1);
                wa = wp[4]; wb = wp[5];
                o2 = fmaf(wa.x,zq[0],o2); o2 = fmaf(wa.y,zq[1],o2); o2 = fmaf(wa.z,zq[2],o2); o2 = fmaf(wa.w,zq[3],o2);
                o2 = fmaf(wb.x,zq[4],o2); o2 = fmaf(wb.y,zq[5],o2); o2 = fmaf(wb.z,zq[6],o2); o2 = fmaf(wb.w,zq[7],o2);
                wa = wp[6]; wb = wp[7];
                o3 = fmaf(wa.x,zq[0],o3); o3 = fmaf(wa.y,zq[1],o3); o3 = fmaf(wa.z,zq[2],o3); o3 = fmaf(wa.w,zq[3],o3);
                o3 = fmaf(wb.x,zq[4],o3); o3 = fmaf(wb.y,zq[5],o3); o3 = fmaf(wb.z,zq[6],o3); o3 = fmaf(wb.w,zq[7],o3);
                r4.x -= o0; r4.y -= o1; r4.z -= o2; r4.w -= o3;
                *(float4*)(rp + k) = r4;
            }
        }
        __syncthreads();
    }

    // ---- epilogue: z_q = z(stored) - residual_final ----
    {
        const float* rp = s_res + tok * RES_STRIDE + dbase;
#pragma unroll 4
        for (int k = 0; k < 64; k += 4) {
            float4 r4 = *(const float4*)(rp + k);
            float z0 = zo[(size_t)(k + 0) * TLEN];
            float z1 = zo[(size_t)(k + 1) * TLEN];
            float z2 = zo[(size_t)(k + 2) * TLEN];
            float z3 = zo[(size_t)(k + 3) * TLEN];
            zo[(size_t)(k + 0) * TLEN] = z0 - r4.x;
            zo[(size_t)(k + 1) * TLEN] = z1 - r4.y;
            zo[(size_t)(k + 2) * TLEN] = z2 - r4.z;
            zo[(size_t)(k + 3) * TLEN] = z3 - r4.w;
        }
    }

    // ---- scalar reductions (KL, losses) ----
#pragma unroll
    for (int off = 16; off >= 1; off >>= 1) {
        klacc   += __shfl_xor(klacc,   off, 32);
        lossacc += __shfl_xor(lossacc, off, 32);
    }
    if (ln == 0) s_red[wv] = klacc;
    __syncthreads();
    if (tid == 0) {
        float s = 0.f;
        for (int w = 0; w < 8; w++) s += s_red[w];
        atomicAdd(out + KL_OFF, s * (1.0f / ((float)BB * TLEN)));
    }
    __syncthreads();
    if (ln == 0) s_red[wv] = lossacc;
    __syncthreads();
    if (tid == 0) {
        float s = 0.f;
        for (int w = 0; w < 8; w++) s += s_red[w];
        float scaled = s * (1.0f / ((float)BB * CD * TLEN * NC));
        atomicAdd(out + CM_OFF,  scaled);
        atomicAdd(out + CBL_OFF, scaled);
    }
}

extern "C" void kernel_launch(void* const* d_in, const int* in_sizes, int n_in,
                              void* d_out, int out_size, void* d_ws, size_t ws_size,
                              hipStream_t stream) {
    (void)in_sizes; (void)n_in; (void)out_size; (void)ws_size;
    const float* x      = (const float*)d_in[0];
    const float* noise  = (const float*)d_in[1];
    const float* in_w   = (const float*)d_in[2];
    const float* in_b   = (const float*)d_in[3];
    const float* cb     = (const float*)d_in[4];
    const float* out_w  = (const float*)d_in[5];
    const float* out_b  = (const float*)d_in[6];
    float* out = (float*)d_out;

    rvq_zero_scalars<<<1, 64, 0, stream>>>(out);
    rvq_norm_cb<<<(NC * CS + 255) / 256, 256, 0, stream>>>(cb, (uint4*)d_ws);
    rvq_main<<<BB * TLEN / TILE, NTHREADS, 0, stream>>>(
        x, noise, in_w, in_b, cb, out_w, out_b, (const uint4*)d_ws, out);
}
